// SSM_28716151341399
// MI455X (gfx1250) — compile-verified, matrix operand order checked by reading, not run on hardware
//
#include <hip/hip_runtime.h>
#include <hip/hip_bf16.h>
#include <math.h>

// ---------------- model dims (compile-time, from reference) ----------------
#define BB   4
#define LL   1024
#define DIN  64
#define DD   512
#define NN   16
#define RR   32
#define MT   4096          // B*L rows of every activation matrix
#define DOUT 10
#define XDW  (RR + 2 * NN) // 64

typedef __attribute__((ext_vector_type(16))) _Float16 v16h;
typedef __attribute__((ext_vector_type(8)))  _Float16 v8h;
typedef __attribute__((ext_vector_type(4)))  _Float16 v4h;
typedef __attribute__((ext_vector_type(8)))  float    v8f;

__device__ __forceinline__ float gelu_exact(float x) {
    return 0.5f * x * (1.0f + erff(x * 0.70710678118654752f));
}
__device__ __forceinline__ float softplus_f(float x) {
    return (x > 20.0f) ? x : log1pf(__expf(x));
}

// ---------------------------------------------------------------------------
// fp32 -> f16 conversion (optionally fused exact GELU), 4 elems/thread.
// All tensor sizes here are multiples of 4.
// ---------------------------------------------------------------------------
__global__ __launch_bounds__(256)
void ssm_cvt_f16(const float* __restrict__ src, _Float16* __restrict__ dst,
                 int n, int gelu)
{
    const int i = (blockIdx.x * 256 + threadIdx.x) * 4;
    if (i >= n) return;
    float4 v = *(const float4*)(src + i);
    if (gelu) {
        v.x = gelu_exact(v.x); v.y = gelu_exact(v.y);
        v.z = gelu_exact(v.z); v.w = gelu_exact(v.w);
    }
    v4h o;
    o[0] = (_Float16)v.x; o[1] = (_Float16)v.y;
    o[2] = (_Float16)v.z; o[3] = (_Float16)v.w;
    *(v4h*)(dst + i) = o;
}

// ---------------------------------------------------------------------------
// WMMA GEMM on f16 operands:
//   C[m][n] = act_out( sum_k A[m][k]*W[n][k] + bias[n] + res[m][n] )
// Block = 128 threads = 4 waves; each wave owns a 32x32 output (2x2 WMMA
// tiles -> 4 v_wmma per K-step); block tile = 64x64; K stepped by 32.
// Fragment loads are two global_load_b128 per 16x32 operand tile.
// ---------------------------------------------------------------------------
__device__ __forceinline__ v16h frag_ld(const _Float16* __restrict__ row, int koff)
{
    v8h lo = *(const v8h*)(row + koff);
    v8h hi = *(const v8h*)(row + koff + 16);
    return __builtin_shufflevector(lo, hi, 0, 1, 2, 3, 4, 5, 6, 7,
                                           8, 9, 10, 11, 12, 13, 14, 15);
}

__global__ __launch_bounds__(128)
void ssm_wmma_gemm16(const _Float16* __restrict__ A, int lda,
                     const _Float16* __restrict__ W, int ldw,
                     const float* __restrict__ bias,
                     const float* __restrict__ res,
                     float* __restrict__ C, int ldc,
                     int K, int softplus_out)
{
    const int lane = threadIdx.x & 31;
    const int wave = threadIdx.x >> 5;
    const int m0  = blockIdx.y * 64 + (wave >> 1) * 32;
    const int n0  = blockIdx.x * 64 + (wave & 1) * 32;
    const int l15 = lane & 15;
    const int kh  = lane >> 4;

    const _Float16* __restrict__ a0p = A + (size_t)(m0 + l15) * lda;
    const _Float16* __restrict__ a1p = A + (size_t)(m0 + 16 + l15) * lda;
    const _Float16* __restrict__ b0p = W + (size_t)(n0 + l15) * ldw;
    const _Float16* __restrict__ b1p = W + (size_t)(n0 + 16 + l15) * ldw;

    v8f acc00 = {}, acc01 = {}, acc10 = {}, acc11 = {};
    for (int kb = 0; kb < K; kb += 32) {
        const int koff = kb + (kh << 3);
        const v16h a0 = frag_ld(a0p, koff);
        const v16h a1 = frag_ld(a1p, koff);
        const v16h b0 = frag_ld(b0p, koff);
        const v16h b1 = frag_ld(b1p, koff);
        if (kb + 32 < K) {                 // pull next K-tile toward the WGP
            __builtin_prefetch(a0p + kb + 32, 0, 1);
            __builtin_prefetch(b0p + kb + 32, 0, 1);
        }
        acc00 = __builtin_amdgcn_wmma_f32_16x16x32_f16(false, a0, false, b0,
                                                       (short)0, acc00, false, false);
        acc01 = __builtin_amdgcn_wmma_f32_16x16x32_f16(false, a0, false, b1,
                                                       (short)0, acc01, false, false);
        acc10 = __builtin_amdgcn_wmma_f32_16x16x32_f16(false, a1, false, b0,
                                                       (short)0, acc10, false, false);
        acc11 = __builtin_amdgcn_wmma_f32_16x16x32_f16(false, a1, false, b1,
                                                       (short)0, acc11, false, false);
    }

    auto epilogue = [&](const v8f& acc, int mbase, int nn) {
        const float bval = bias ? bias[nn] : 0.0f;
#pragma unroll
        for (int r = 0; r < 8; ++r) {
            const int mm = mbase + r + 8 * kh;     // ISA C/D layout
            float v = acc[r] + bval;
            if (res) v += res[(size_t)mm * ldc + nn];
            if (softplus_out) v = softplus_f(v);
            C[(size_t)mm * ldc + nn] = v;
        }
    };
    const int nnA = n0 + l15;
    epilogue(acc00, m0,      nnA);
    epilogue(acc01, m0,      nnA + 16);
    epilogue(acc10, m0 + 16, nnA);
    epilogue(acc11, m0 + 16, nnA + 16);
}

// ---------------------------------------------------------------------------
// Selective scan: one thread per (b,d) channel, 16-wide state in registers.
// B_t / C_t (32 floats, shared across all d of a batch) staged in LDS.
// grid = (DD/256, BB), block = 256
// ---------------------------------------------------------------------------
__global__ __launch_bounds__(256)
void ssm_scan(const float* __restrict__ h,      // (B,L,D)
              const float* __restrict__ xdbl,   // (B,L,R+2N)
              const float* __restrict__ delta,  // (B,L,D)
              const float* __restrict__ Alog,   // (D,N)
              const float* __restrict__ Dp,     // (D)
              float* __restrict__ y)            // (B,L,D)
{
    __shared__ float sBC[2 * NN];
    const int d = blockIdx.x * 256 + threadIdx.x;
    const int b = blockIdx.y;

    float Aa[NN], s[NN];
#pragma unroll
    for (int n = 0; n < NN; ++n) {
        Aa[n] = -__expf(Alog[d * NN + n]);
        s[n]  = 0.0f;
    }
    const float Dd = Dp[d];
    const size_t baseBL = (size_t)b * LL;

    for (int t = 0; t < LL; ++t) {
        __syncthreads();
        if (threadIdx.x < 2 * NN)
            sBC[threadIdx.x] = xdbl[(baseBL + t) * XDW + RR + threadIdx.x];
        __syncthreads();

        const float dlt = delta[(baseBL + t) * DD + d];
        const float hv  = h[(baseBL + t) * DD + d];
        const float dh  = dlt * hv;
        float acc = 0.0f;
#pragma unroll
        for (int n = 0; n < NN; ++n) {
            const float dA = __expf(dlt * Aa[n]);
            s[n] = dA * s[n] + dh * sBC[n];
            acc += s[n] * sBC[NN + n];
        }
        y[(baseBL + t) * DD + d] = acc + hv * Dd;
    }
}

// mean over L: one thread per (b,d)
__global__ __launch_bounds__(256)
void ssm_mean(const float* __restrict__ h, float* __restrict__ mean)
{
    const int idx = blockIdx.x * blockDim.x + threadIdx.x;   // b*DD + d
    const int b = idx >> 9, d = idx & (DD - 1);
    float s = 0.0f;
    for (int t = 0; t < LL; ++t)
        s += h[((size_t)b * LL + t) * DD + d];
    mean[idx] = s * (1.0f / (float)LL);
}

// output projection: 40 dots of length 512
__global__ __launch_bounds__(64)
void ssm_outproj(const float* __restrict__ mean,
                 const float* __restrict__ ow, const float* __restrict__ ob,
                 float* __restrict__ out)
{
    const int i = threadIdx.x;
    if (i >= BB * DOUT) return;
    const int b = i / DOUT, o = i % DOUT;
    float s = ob[o];
    for (int k = 0; k < DD; ++k)
        s += mean[b * DD + k] * ow[o * DD + k];
    out[i] = s;
}

// ---------------------------------------------------------------------------
extern "C" void kernel_launch(void* const* d_in, const int* in_sizes, int n_in,
                              void* d_out, int out_size, void* d_ws, size_t ws_size,
                              hipStream_t stream)
{
    const float* x      = (const float*)d_in[0];
    const float* W_in   = (const float*)d_in[1];
    const float* b_in   = (const float*)d_in[2];
    const float* mix1_w = (const float*)d_in[3];
    const float* mix1_b = (const float*)d_in[4];
    const float* mix2_w = (const float*)d_in[5];
    const float* mix2_b = (const float*)d_in[6];
    const float* out_w  = (const float*)d_in[7];
    const float* out_b  = (const float*)d_in[8];
    const float* m1_xp  = (const float*)d_in[9];
    const float* m1_dtw = (const float*)d_in[10];
    const float* m1_dtb = (const float*)d_in[11];
    const float* m1_Al  = (const float*)d_in[12];
    const float* m1_D   = (const float*)d_in[13];
    const float* m2_xp  = (const float*)d_in[14];
    const float* m2_dtw = (const float*)d_in[15];
    const float* m2_dtb = (const float*)d_in[16];
    const float* m2_Al  = (const float*)d_in[17];
    const float* m2_D   = (const float*)d_in[18];
    float* out = (float*)d_out;

    // ---- workspace layout ----
    float* ws = (float*)d_ws;
    float* h1 = ws;                                   // MT*DD
    float* h2 = h1 + (size_t)MT * DD;                 // MT*DD
    float* yb = h2 + (size_t)MT * DD;                 // MT*DD
    float* xd = yb + (size_t)MT * DD;                 // MT*XDW
    float* dl = xd + (size_t)MT * XDW;                // MT*DD
    float* mn = dl + (size_t)MT * DD;                 // BB*DD
    _Float16* f16 = (_Float16*)(mn + (size_t)BB * DD);
    _Float16* act16  = f16;                           // MT*DD   (reused)
    _Float16* x16    = act16  + (size_t)MT * DD;      // MT*DIN
    _Float16* xd16   = x16    + (size_t)MT * DIN;     // MT*XDW
    _Float16* wi16   = xd16   + (size_t)MT * XDW;     // DD*DIN
    _Float16* xp1_16 = wi16   + (size_t)DD * DIN;     // XDW*DD
    _Float16* xp2_16 = xp1_16 + (size_t)XDW * DD;
    _Float16* dtw1_16= xp2_16 + (size_t)XDW * DD;     // DD*RR
    _Float16* dtw2_16= dtw1_16+ (size_t)DD * RR;
    _Float16* mx1_16 = dtw2_16+ (size_t)DD * RR;      // DD*DD
    _Float16* mx2_16 = mx1_16 + (size_t)DD * DD;

    auto cvt = [&](const float* s, _Float16* d, int n, int g) {
        ssm_cvt_f16<<<(n / 4 + 255) / 256, 256, 0, stream>>>(s, d, n, g);
    };

    const dim3 blk(128);
    const dim3 gFull(DD / 64, MT / 64);     // N=512
    const dim3 gProj(XDW / 64, MT / 64);    // N=64
    const dim3 gScan(DD / 256, BB);

    // ---- one-time (per launch) weight conversions ----
    cvt(W_in,   wi16,    DD * DIN, 0);
    cvt(m1_xp,  xp1_16,  XDW * DD, 0);
    cvt(m1_dtw, dtw1_16, DD * RR,  0);
    cvt(mix1_w, mx1_16,  DD * DD,  0);
    cvt(m2_xp,  xp2_16,  XDW * DD, 0);
    cvt(m2_dtw, dtw2_16, DD * RR,  0);
    cvt(mix2_w, mx2_16,  DD * DD,  0);

    // h1 = x @ W_in^T + b_in
    cvt(x, x16, MT * DIN, 0);
    ssm_wmma_gemm16<<<gFull, blk, 0, stream>>>(x16, DIN, wi16, DIN, b_in, nullptr,
                                               h1, DD, DIN, 0);
    // ---- mamba block 1 ----
    cvt(h1, act16, MT * DD, 0);
    ssm_wmma_gemm16<<<gProj, blk, 0, stream>>>(act16, DD, xp1_16, DD, nullptr, nullptr,
                                               xd, XDW, DD, 0);
    cvt(xd, xd16, MT * XDW, 0);
    ssm_wmma_gemm16<<<gFull, blk, 0, stream>>>(xd16, XDW, dtw1_16, RR, m1_dtb, nullptr,
                                               dl, DD, RR, 1);
    ssm_scan<<<gScan, 256, 0, stream>>>(h1, xd, dl, m1_Al, m1_D, yb);
    // h2 = gelu(yb) @ mix1^T + mix1_b + h1
    cvt(yb, act16, MT * DD, 1);
    ssm_wmma_gemm16<<<gFull, blk, 0, stream>>>(act16, DD, mx1_16, DD, mix1_b, h1,
                                               h2, DD, DD, 0);
    // ---- mamba block 2 ----
    cvt(h2, act16, MT * DD, 0);
    ssm_wmma_gemm16<<<gProj, blk, 0, stream>>>(act16, DD, xp2_16, DD, nullptr, nullptr,
                                               xd, XDW, DD, 0);
    cvt(xd, xd16, MT * XDW, 0);
    ssm_wmma_gemm16<<<gFull, blk, 0, stream>>>(xd16, XDW, dtw2_16, RR, m2_dtb, nullptr,
                                               dl, DD, RR, 1);
    ssm_scan<<<gScan, 256, 0, stream>>>(h2, xd, dl, m2_Al, m2_D, yb);
    // h1 = gelu(yb) @ mix2^T + mix2_b + h2
    cvt(yb, act16, MT * DD, 1);
    ssm_wmma_gemm16<<<gFull, blk, 0, stream>>>(act16, DD, mx2_16, DD, mix2_b, h2,
                                               h1, DD, DD, 0);
    // pool + head
    ssm_mean<<<(BB * DD) / 256, 256, 0, stream>>>(h1, mn);
    ssm_outproj<<<1, 64, 0, stream>>>(mn, out_w, out_b, out);

    (void)in_sizes; (void)n_in; (void)out_size; (void)ws_size;
}
